// EvoformerBlock_34187939676341
// MI455X (gfx1250) — compile-verified, matrix operand order checked by reading, not run on hardware
//
#include <hip/hip_runtime.h>

// ---------------------------------------------------------------------------
// Evoformer block for gfx1250 (MI455X). All matmuls via v_wmma_f32_16x16x32_bf16
// (bf16 inputs, f32 accumulate). f32 everywhere else (LN stats, softmax, residuals).
// ---------------------------------------------------------------------------

typedef __attribute__((ext_vector_type(16))) __bf16 v16bf;
typedef __attribute__((ext_vector_type(8)))  float  v8f;

#define LN_EPS 1e-5f

// Native bf16 converts (let the backend use v_cvt*bf16* instructions).
__device__ __forceinline__ unsigned short f2bf(float x) {
  union { __bf16 b; unsigned short u; } v; v.b = (__bf16)x; return v.u;
}
__device__ __forceinline__ float bf2f(unsigned short h) {
  union { float f; unsigned u; } v; v.u = ((unsigned)h) << 16; return v.f;
}
__device__ __forceinline__ unsigned pack2(float lo, float hi) {
  union { __bf16 b[2]; unsigned u; } v; v.b[0] = (__bf16)lo; v.b[1] = (__bf16)hi;
  return v.u;
}

union FragU { v16bf v; unsigned u[8]; };

// bf16 fragment: 16 "rows" x 32 reduction elems, from reduction-contiguous bf16
// storage. Matches CDNA5 16-bit A(16x32) and B(32x16) VGPR layouts:
// lanes 0-15: row=lane,  K = {k0..k0+7, k0+16..k0+23}
// lanes 16-31: row=lane-16, K shifted by +8.
__device__ __forceinline__ v16bf load_frag_bf(const unsigned short* base, int row0,
                                              int rowStride, int k0, int rowMax) {
  int lane = threadIdx.x & 31;
  int r  = row0 + (lane & 15);
  int kb = k0 + ((lane >> 4) << 3);
  FragU f;
  if (r < rowMax) {
    const unsigned short* p = base + (size_t)r * rowStride + kb;
#pragma unroll
    for (int i = 0; i < 4; ++i) f.u[i]     = *(const unsigned*)(p + 2 * i);
#pragma unroll
    for (int i = 0; i < 4; ++i) f.u[4 + i] = *(const unsigned*)(p + 16 + 2 * i);
  } else {
#pragma unroll
    for (int i = 0; i < 8; ++i) f.u[i] = 0u;
  }
  return f.v;
}

// Same fragment but sourced from f32 storage (activations), converting to bf16.
__device__ __forceinline__ v16bf load_frag_f32(const float* base, int row0,
                                               int rowStride, int k0, int rowMax) {
  int lane = threadIdx.x & 31;
  int r  = row0 + (lane & 15);
  int kb = k0 + ((lane >> 4) << 3);
  FragU f;
  if (r < rowMax) {
    const float* p = base + (size_t)r * rowStride + kb;
    float4 a0 = *(const float4*)(p);
    float4 a1 = *(const float4*)(p + 4);
    float4 b0 = *(const float4*)(p + 16);
    float4 b1 = *(const float4*)(p + 20);
    f.u[0] = pack2(a0.x, a0.y); f.u[1] = pack2(a0.z, a0.w);
    f.u[2] = pack2(a1.x, a1.y); f.u[3] = pack2(a1.z, a1.w);
    f.u[4] = pack2(b0.x, b0.y); f.u[5] = pack2(b0.z, b0.w);
    f.u[6] = pack2(b1.x, b1.y); f.u[7] = pack2(b1.z, b1.w);
  } else {
#pragma unroll
    for (int i = 0; i < 8; ++i) f.u[i] = 0u;
  }
  return f.v;
}

// --------------------------- weight prep ------------------------------------
// W[K][N] f32 (row-major) -> Wt[N][K] bf16 (reduction-contiguous)
__global__ void wprep_kernel(const float* __restrict__ w, unsigned short* __restrict__ wt,
                             int K, int N) {
  int idx = blockIdx.x * 256 + threadIdx.x;
  if (idx >= K * N) return;
  int n = idx / K, k = idx % K;
  wt[idx] = f2bf(w[(size_t)k * N + n]);
}

// --------------------------- generic linear GEMM ----------------------------
// C[M][N] = epilogue( A[M][K] @ Wt^T + bias ); flags: 1=ReLU, 2=*sigmoid(gate), 4=+res
// Wave tile: 16(M) x 32(N) -> one A fragment feeds two WMMAs.
__global__ void gemm_kernel(const float* __restrict__ A, const unsigned short* __restrict__ Wt,
                            const float* __restrict__ bias, float* __restrict__ C,
                            int M, int N, int K, int flags,
                            const float* __restrict__ gate, const float* __restrict__ res) {
  int wave = threadIdx.x >> 5;
  int m0 = blockIdx.x * 128 + wave * 16;
  int n0 = blockIdx.y * 32;
  v8f acc0 = {}, acc1 = {};
  for (int k0 = 0; k0 < K; k0 += 32) {
    __builtin_prefetch(A + (size_t)m0 * K + k0 + 32, 0, 0);
    v16bf a  = load_frag_f32(A,  m0, K, k0, M);
    v16bf b0 = load_frag_bf(Wt, n0,      K, k0, N);
    v16bf b1 = load_frag_bf(Wt, n0 + 16, K, k0, N);
    acc0 = __builtin_amdgcn_wmma_f32_16x16x32_bf16(false, a, false, b0, (short)0, acc0,
                                                   false, false);
    acc1 = __builtin_amdgcn_wmma_f32_16x16x32_bf16(false, a, false, b1, (short)0, acc1,
                                                   false, false);
  }
  int lane = threadIdx.x & 31;
  int half = lane >> 4;
#pragma unroll
  for (int t = 0; t < 2; ++t) {
    int col = n0 + t * 16 + (lane & 15);
    if (col < N) {
      v8f acc = t ? acc1 : acc0;
      float bv = bias ? bias[col] : 0.0f;
#pragma unroll
      for (int r = 0; r < 8; ++r) {
        int m = m0 + r + half * 8;
        if (m < M) {
          float v = acc[r] + bv;
          if (flags & 1) v = fmaxf(v, 0.0f);
          size_t o = (size_t)m * N + col;
          if (flags & 2) v *= 1.0f / (1.0f + __expf(-gate[o]));
          if (flags & 4) v += res[o];
          C[o] = v;
        }
      }
    }
  }
}

// --------------------------- channel-batched GEMM (triangle mul) -------------
// A[c][L][L], B[c][L][L] bf16 (both reduction-contiguous); out[i][j][c] f32.
// Wave tile 16x32 as above.
__global__ void bgemm_kernel(const unsigned short* __restrict__ A,
                             const unsigned short* __restrict__ B,
                             float* __restrict__ out, int C, int L) {
  int c    = blockIdx.z;
  int wave = threadIdx.x >> 5;
  int m0 = blockIdx.x * 128 + wave * 16;
  int n0 = blockIdx.y * 32;
  const unsigned short* Ab = A + (size_t)c * L * L;
  const unsigned short* Bb = B + (size_t)c * L * L;
  v8f acc0 = {}, acc1 = {};
  for (int k0 = 0; k0 < L; k0 += 32) {
    v16bf a  = load_frag_bf(Ab, m0, L, k0, L);
    v16bf b0 = load_frag_bf(Bb, n0,      L, k0, L);
    v16bf b1 = load_frag_bf(Bb, n0 + 16, L, k0, L);
    acc0 = __builtin_amdgcn_wmma_f32_16x16x32_bf16(false, a, false, b0, (short)0, acc0,
                                                   false, false);
    acc1 = __builtin_amdgcn_wmma_f32_16x16x32_bf16(false, a, false, b1, (short)0, acc1,
                                                   false, false);
  }
  int lane = threadIdx.x & 31;
  int half = lane >> 4;
#pragma unroll
  for (int t = 0; t < 2; ++t) {
    int n = n0 + t * 16 + (lane & 15);
    v8f acc = t ? acc1 : acc0;
#pragma unroll
    for (int r = 0; r < 8; ++r) {
      int m = m0 + r + half * 8;
      out[((size_t)m * L + n) * C + c] = acc[r];
    }
  }
}

// --------------------------- fused attention --------------------------------
// One (batch,head) per workgroup. Q/K/V staged in LDS as bf16 padded to 32 dims,
// scores (bf16) in LDS, in-LDS softmax, then A@V. Covers row/col/triangle attn.
// element address: base + b1*bStride + s*sStride + h*hd + d
__global__ void attn_kernel(const float* __restrict__ Q, const float* __restrict__ Kp,
                            const float* __restrict__ V, float* __restrict__ O,
                            const float* __restrict__ bias, int H, int S, int hd,
                            int bStride, int sStride, float scale) {
  extern __shared__ unsigned short sm[];
  unsigned short* Qs = sm;                 // [S][32]
  unsigned short* Ks = Qs + S * 32;        // [S][32]
  unsigned short* Vt = Ks + S * 32;        // [32][S]  (dims x keys)
  unsigned short* Sc = Vt + 32 * S;        // [S][S]   scores -> probs (bf16)

  int b1 = blockIdx.x, h = blockIdx.y;
  size_t qoff = (size_t)b1 * bStride + (size_t)h * hd;
  const float* qb = Q  + qoff;
  const float* kb = Kp + qoff;
  const float* vb = V  + qoff;
  float*       ob = O  + qoff;

  for (int s = threadIdx.x; s < S; s += blockDim.x) {
    const float* qr = qb + (size_t)s * sStride;
    const float* kr = kb + (size_t)s * sStride;
    const float* vr = vb + (size_t)s * sStride;
    unsigned* qd = (unsigned*)(Qs + s * 32);
    unsigned* kd = (unsigned*)(Ks + s * 32);
    for (int d = 0; d < 32; d += 2) {
      float q0 = (d < hd) ? qr[d] : 0.0f, q1 = (d + 1 < hd) ? qr[d + 1] : 0.0f;
      float k0 = (d < hd) ? kr[d] : 0.0f, k1 = (d + 1 < hd) ? kr[d + 1] : 0.0f;
      qd[d >> 1] = pack2(q0, q1);
      kd[d >> 1] = pack2(k0, k1);
      Vt[d * S + s]       = (d < hd)     ? f2bf(vr[d])     : (unsigned short)0;
      Vt[(d + 1) * S + s] = (d + 1 < hd) ? f2bf(vr[d + 1]) : (unsigned short)0;
    }
  }
  __syncthreads();

  int wave = threadIdx.x >> 5;
  int lane = threadIdx.x & 31;
  int half = lane >> 4;
  int nT = S / 16;

  // scores: one 16x16x32 WMMA per tile (hd padded to 32 with zeros)
  for (int t = wave; t < nT * nT; t += 8) {
    int qt = t / nT, kt = t % nT;
    v16bf a = load_frag_bf(Qs, qt * 16, 32, 0, S);
    v16bf b = load_frag_bf(Ks, kt * 16, 32, 0, S);
    v8f  c = {};
    c = __builtin_amdgcn_wmma_f32_16x16x32_bf16(false, a, false, b, (short)0, c,
                                                false, false);
    int n = kt * 16 + (lane & 15);
#pragma unroll
    for (int r = 0; r < 8; ++r) {
      int m = qt * 16 + r + half * 8;
      float v = c[r] * scale;
      if (bias) v += bias[((size_t)m * S + n) * H + h];
      Sc[(size_t)m * S + n] = f2bf(v);
    }
  }
  __syncthreads();

  // softmax per row (mask is all-true in this problem)
  for (int rr = threadIdx.x; rr < S; rr += blockDim.x) {
    unsigned short* row = Sc + (size_t)rr * S;
    float mx = -3.0e38f;
    for (int j = 0; j < S; ++j) mx = fmaxf(mx, bf2f(row[j]));
    float sum = 0.0f;
    for (int j = 0; j < S; ++j) sum += __expf(bf2f(row[j]) - mx);
    float inv = 1.0f / sum;
    for (int j = 0; j < S; ++j) row[j] = f2bf(__expf(bf2f(row[j]) - mx) * inv);
  }
  __syncthreads();

  // O = P @ V
  int nNT = (hd + 15) / 16;
  for (int t = wave; t < nT * nNT; t += 8) {
    int qt = t / nNT, nt = t % nNT;
    v8f acc = {};
    for (int kc = 0; kc < S; kc += 32) {
      v16bf a = load_frag_bf(Sc, qt * 16, S, kc, S);
      v16bf b = load_frag_bf(Vt, nt * 16, S, kc, 32);
      acc = __builtin_amdgcn_wmma_f32_16x16x32_bf16(false, a, false, b, (short)0, acc,
                                                    false, false);
    }
    int col = nt * 16 + (lane & 15);
    if (col < hd) {
#pragma unroll
      for (int r = 0; r < 8; ++r) {
        int m = qt * 16 + r + half * 8;
        ob[(size_t)m * sStride + col] = acc[r];
      }
    }
  }
}

// --------------------------- LayerNorm (one wave32 per row) ------------------
__global__ void ln_kernel(const float* __restrict__ x, const float* __restrict__ res,
                          const float* __restrict__ g, const float* __restrict__ b,
                          float* __restrict__ out, int rows, int D) {
  int wave = threadIdx.x >> 5, lane = threadIdx.x & 31;
  int row = blockIdx.x * 8 + wave;
  if (row >= rows) return;
  const float* xr = x + (size_t)row * D;
  const float* rr = res ? res + (size_t)row * D : nullptr;
  int per = D >> 5;                      // 2 (D=64) or 4 (D=128)
  float vals[4];
  float s = 0.0f;
  for (int i = 0; i < per; ++i) {
    int c = lane + 32 * i;
    float v = xr[c] + (rr ? rr[c] : 0.0f);
    vals[i] = v; s += v;
  }
  for (int o = 16; o >= 1; o >>= 1) s += __shfl_xor(s, o, 32);
  float mean = s / (float)D;
  float vs = 0.0f;
  for (int i = 0; i < per; ++i) { float d = vals[i] - mean; vs += d * d; }
  for (int o = 16; o >= 1; o >>= 1) vs += __shfl_xor(vs, o, 32);
  float inv = rsqrtf(vs / (float)D + LN_EPS);
  for (int i = 0; i < per; ++i) {
    int c = lane + 32 * i;
    out[(size_t)row * D + c] = (vals[i] - mean) * inv * g[c] + b[c];
  }
}

// --------------------------- misc elementwise --------------------------------
__global__ void pooled_kernel(const float* __restrict__ msa, const unsigned char* __restrict__ mask,
                              float* __restrict__ out, int Nn, int L, int D) {
  int idx = blockIdx.x * 256 + threadIdx.x;
  if (idx >= L * D) return;
  int l = idx / D, d = idx % D;
  float s = 0.0f, c = 0.0f;
  for (int n = 0; n < Nn; ++n) {
    float m = mask[(size_t)n * L + l] ? 1.0f : 0.0f;
    s += msa[((size_t)n * L + l) * D + d] * m;
    c += m;
  }
  out[idx] = s / (c + 1e-8f);
}

__global__ void outer_kernel(const float* __restrict__ pin, const float* __restrict__ a,
                             const float* __restrict__ b, float* __restrict__ out,
                             int L, int C) {
  size_t idx = (size_t)blockIdx.x * 256 + threadIdx.x;
  if (idx >= (size_t)L * L * C) return;
  int c = (int)(idx % C);
  size_t ij = idx / C;
  int j = (int)(ij % L), i = (int)(ij / L);
  out[idx] = pin[idx] + a[(size_t)i * C + c] * b[(size_t)j * C + c];
}

// f32 [L][L][C] -> bf16 channel-major [C][x][y]; swap=0: (x,y)=(i,j); swap=1: (x,y)=(j,i)
__global__ void tcvt_kernel(const float* __restrict__ in, unsigned short* __restrict__ out,
                            int L, int C, int swap) {
  size_t idx = (size_t)blockIdx.x * 256 + threadIdx.x;
  if (idx >= (size_t)L * L * C) return;
  int y = (int)(idx % L);
  int x = (int)((idx / L) % L);
  int c = (int)(idx / ((size_t)L * L));
  size_t src = swap ? ((size_t)y * L + x) * C + c : ((size_t)x * L + y) * C + c;
  out[idx] = f2bf(in[src]);
}

// ---------------------------------------------------------------------------
extern "C" void kernel_launch(void* const* d_in, const int* in_sizes, int n_in,
                              void* d_out, int out_size, void* d_ws, size_t ws_size,
                              hipStream_t stream) {
  (void)in_sizes; (void)n_in; (void)out_size; (void)ws_size;
  const int Nseq = 128, L = 256, DM = 64, DP = 128, RH = 8, CHD = 8, THD = 4;
  const size_t NL = (size_t)Nseq * L, LL = (size_t)L * L;
  const size_t MSA_E = NL * DM;
  const int NLi = (int)NL, LLi = (int)LL;

  const float* msa_in  = (const float*)d_in[0];
  const float* pair_in = (const float*)d_in[1];
  const unsigned char* mask = (const unsigned char*)d_in[2];
  auto B = [&](int i) { return (const float*)d_in[i]; };

  // ---- workspace arena (peak ~220 MB) ----
  char* basep = (char*)d_ws; size_t off = 0;
  auto arena = [&](size_t bytes) -> void* {
    void* p = basep + off; off = (off + bytes + 255) & ~(size_t)255; return p;
  };
  float* bufMsa  = (float*)arena(MSA_E * 4);
  float* bufPair = (float*)arena(LL * DP * 4);
  float* tA = (float*)arena(LL * DP * 4);
  float* tB = (float*)arena(LL * DP * 4);
  float* tC = (float*)arena(LL * DP * 4);
  float* tD = (float*)arena(LL * DP * 4);
  unsigned short* bfA = (unsigned short*)arena(LL * DP * 2);
  unsigned short* bfB = (unsigned short*)arena(LL * DP * 2);
  float* hid    = (float*)arena((size_t)8192 * 512 * 4);
  float* pb     = (float*)arena(LL * RH * 4);
  float* pooled = (float*)arena((size_t)L * DM * 4);
  float* oa     = (float*)arena((size_t)L * DP * 4);
  float* ob2    = (float*)arena((size_t)L * DP * 4);

  // ---- pre-transpose + bf16-convert all weights ----
  auto wprepf = [&](int idx, int K, int Nn) -> const unsigned short* {
    unsigned short* dst = (unsigned short*)arena((size_t)K * Nn * 2);
    int tot = K * Nn;
    wprep_kernel<<<dim3((tot + 255) / 256), dim3(256), 0, stream>>>(
        (const float*)d_in[idx], dst, K, Nn);
    return dst;
  };
  // input flattening: setup_inputs() insertion order; each lin = {w,b}, ln = {g,b}
  const unsigned short *w_rq = wprepf(3, 64, 64),  *w_rk = wprepf(5, 64, 64),
                       *w_rv = wprepf(7, 64, 64),  *w_ro = wprepf(9, 64, 64),
                       *w_rg = wprepf(11, 64, 64), *w_rpb = wprepf(13, 128, 8);
  const unsigned short *w_cq = wprepf(15, 64, 64), *w_ck = wprepf(17, 64, 64),
                       *w_cv = wprepf(19, 64, 64), *w_co = wprepf(21, 64, 64),
                       *w_cg = wprepf(23, 64, 64);
  const unsigned short *w_mt1 = wprepf(31, 64, 256), *w_mt2 = wprepf(33, 256, 64);
  const unsigned short *w_opl = wprepf(35, 64, 128), *w_opr = wprepf(37, 64, 128);
  const unsigned short *w_to_l = wprepf(39, 128, 128), *w_to_r = wprepf(41, 128, 128),
                       *w_to_lg = wprepf(43, 128, 128), *w_to_rg = wprepf(45, 128, 128),
                       *w_to_o = wprepf(47, 128, 128), *w_to_og = wprepf(49, 128, 128);
  const unsigned short *w_ti_l = wprepf(53, 128, 128), *w_ti_r = wprepf(55, 128, 128),
                       *w_ti_lg = wprepf(57, 128, 128), *w_ti_rg = wprepf(59, 128, 128),
                       *w_ti_o = wprepf(61, 128, 128), *w_ti_og = wprepf(63, 128, 128);
  const unsigned short *w_aoq = wprepf(67, 128, 128), *w_aok = wprepf(69, 128, 128),
                       *w_aov = wprepf(71, 128, 128), *w_aoo = wprepf(73, 128, 128),
                       *w_aog = wprepf(75, 128, 128);
  const unsigned short *w_aiq = wprepf(77, 128, 128), *w_aik = wprepf(79, 128, 128),
                       *w_aiv = wprepf(81, 128, 128), *w_aio = wprepf(83, 128, 128),
                       *w_aig = wprepf(85, 128, 128);
  const unsigned short *w_pt1 = wprepf(93, 128, 512), *w_pt2 = wprepf(95, 512, 128);

  auto gemm = [&](const float* A, const unsigned short* W, const float* bias, float* C,
                  int M, int Nn, int K, int flags, const float* gate, const float* res) {
    dim3 g((M + 127) / 128, (Nn + 31) / 32);
    gemm_kernel<<<g, 256, 0, stream>>>(A, W, bias, C, M, Nn, K, flags, gate, res);
  };
  auto lnf = [&](const float* x, const float* res, int gi, int bi, float* outp,
                 int rows, int D) {
    ln_kernel<<<dim3((rows + 7) / 8), 256, 0, stream>>>(x, res, B(gi), B(bi), outp, rows, D);
  };
  (void)hipFuncSetAttribute(reinterpret_cast<const void*>(attn_kernel),
                            hipFuncAttributeMaxDynamicSharedMemorySize, 181 * 1024);
  auto attn = [&](const float* Q, const float* K, const float* V, float* O,
                  const float* bias, int B1, int H, int S, int hd, int bStr, int sStr,
                  float scale) {
    size_t smem = (size_t)(3 * S * 32 + S * S) * 2;
    attn_kernel<<<dim3(B1, H), 256, smem, stream>>>(Q, K, V, O, bias, H, S, hd,
                                                    bStr, sStr, scale);
  };

  float* msaOut  = (float*)d_out;
  float* pairOut = (float*)d_out + MSA_E;

  // ================= MSA stack =================
  // row attention (with pair bias) + gate, residual LN
  gemm(pair_in, w_rpb, B(14), pb, LLi, 8, 128, 0, nullptr, nullptr);
  gemm(msa_in, w_rq, B(4),  tA, NLi, 64, 64, 0, nullptr, nullptr);
  gemm(msa_in, w_rk, B(6),  tB, NLi, 64, 64, 0, nullptr, nullptr);
  gemm(msa_in, w_rv, B(8),  tC, NLi, 64, 64, 0, nullptr, nullptr);
  gemm(msa_in, w_rg, B(12), tD, NLi, 64, 64, 0, nullptr, nullptr);
  attn(tA, tB, tC, hid, pb, Nseq, RH, L, 8, L * DM, DM, 0.35355339f);
  gemm(hid, w_ro, B(10), tB, NLi, 64, 64, 2, tD, nullptr);
  lnf(tB, msa_in, 25, 26, bufMsa, NLi, 64);

  // column attention
  gemm(bufMsa, w_cq, B(16), tA, NLi, 64, 64, 0, nullptr, nullptr);
  gemm(bufMsa, w_ck, B(18), tB, NLi, 64, 64, 0, nullptr, nullptr);
  gemm(bufMsa, w_cv, B(20), tC, NLi, 64, 64, 0, nullptr, nullptr);
  gemm(bufMsa, w_cg, B(24), tD, NLi, 64, 64, 0, nullptr, nullptr);
  attn(tA, tB, tC, hid, nullptr, L, CHD, Nseq, 8, DM, L * DM, 0.35355339f);
  gemm(hid, w_co, B(22), tA, NLi, 64, 64, 2, tD, nullptr);
  lnf(tA, bufMsa, 27, 28, bufMsa, NLi, 64);

  // MSA transition (chunked), write final MSA to d_out
  for (int c = 0; c < 4; ++c) {
    const float* src = bufMsa + (size_t)c * 8192 * DM;
    gemm(src, w_mt1, B(32), hid, 8192, 256, 64, 1, nullptr, nullptr);
    gemm(hid, w_mt2, B(34), tA + (size_t)c * 8192 * DM, 8192, 64, 256, 0, nullptr, nullptr);
  }
  lnf(tA, bufMsa, 29, 30, msaOut, NLi, 64);

  // ================= outer-product mean -> pair =================
  pooled_kernel<<<dim3((L * DM + 255) / 256), 256, 0, stream>>>(msaOut, mask, pooled,
                                                                Nseq, L, DM);
  gemm(pooled, w_opl, B(36), oa,  L, DP, DM, 0, nullptr, nullptr);
  gemm(pooled, w_opr, B(38), ob2, L, DP, DM, 0, nullptr, nullptr);
  {
    size_t tot = LL * DP;
    outer_kernel<<<dim3((unsigned)((tot + 255) / 256)), 256, 0, stream>>>(
        pair_in, oa, ob2, bufPair, L, DP);
  }

  // ================= triangle multiplication =================
  struct TM { const unsigned short *l, *r, *lg, *rg, *o, *og;
              int bl, br, blg, brg, bo, bog, lnG, lnB; };
  TM tmo = { w_to_l, w_to_r, w_to_lg, w_to_rg, w_to_o, w_to_og,
             40, 42, 44, 46, 48, 50, 51, 52 };
  TM tmi = { w_ti_l, w_ti_r, w_ti_lg, w_ti_rg, w_ti_o, w_ti_og,
             54, 56, 58, 60, 62, 64, 65, 66 };
  auto do_trimul = [&](const TM& t, int swapA) {
    gemm(bufPair, t.og, B(t.bog), tA, LLi, DP, DP, 0, nullptr, nullptr);
    gemm(bufPair, t.lg, B(t.blg), tB, LLi, DP, DP, 0, nullptr, nullptr);
    gemm(bufPair, t.l,  B(t.bl),  tC, LLi, DP, DP, 2, tB, nullptr);   // gated left
    gemm(bufPair, t.rg, B(t.brg), tB, LLi, DP, DP, 0, nullptr, nullptr);
    gemm(bufPair, t.r,  B(t.br),  tD, LLi, DP, DP, 2, tB, nullptr);   // gated right
    size_t tot = LL * DP;
    tcvt_kernel<<<dim3((unsigned)((tot + 255) / 256)), 256, 0, stream>>>(tC, bfA, L, DP, swapA);
    tcvt_kernel<<<dim3((unsigned)((tot + 255) / 256)), 256, 0, stream>>>(tD, bfB, L, DP, 1);
    bgemm_kernel<<<dim3(L / 128, L / 32, DP), 256, 0, stream>>>(bfA, bfB, tC, DP, L);
    lnf(tC, nullptr, t.lnG, t.lnB, tD, LLi, DP);
    gemm(tD, t.o, B(t.bo), bufPair, LLi, DP, DP, 2 | 4, tA, bufPair); // pair += upd*sig(og)
  };
  do_trimul(tmo, 0);  // 'out' orientation
  do_trimul(tmi, 1);  // 'in' orientation (transpose absorbed into tcvt swap)

  // ================= triangle attention =================
  auto do_triattn = [&](const unsigned short* wq, int bq, const unsigned short* wk, int bk,
                        const unsigned short* wv, int bv, const unsigned short* wo, int bo,
                        const unsigned short* wg, int bg, int plnG, int plnB) {
    gemm(bufPair, wq, B(bq), tA, LLi, DP, DP, 0, nullptr, nullptr);
    gemm(bufPair, wk, B(bk), tB, LLi, DP, DP, 0, nullptr, nullptr);
    gemm(bufPair, wv, B(bv), tC, LLi, DP, DP, 0, nullptr, nullptr);
    attn(tA, tB, tC, tD, nullptr, L, THD, L, 32, L * DP, DP, 0.17677670f);
    gemm(bufPair, wg, B(bg), tA, LLi, DP, DP, 0, nullptr, nullptr);
    gemm(tD, wo, B(bo), tB, LLi, DP, DP, 2, tA, nullptr);
    lnf(tB, bufPair, plnG, plnB, bufPair, LLi, DP);
  };
  do_triattn(w_aoq, 68, w_aok, 70, w_aov, 72, w_aoo, 74, w_aog, 76, 87, 88); // 'out'
  do_triattn(w_aiq, 78, w_aik, 80, w_aiv, 82, w_aio, 84, w_aig, 86, 89, 90); // 'in'

  // ================= pair transition (chunked), final LN into d_out ========
  for (int c = 0; c < 8; ++c) {
    const float* src = bufPair + (size_t)c * 8192 * DP;
    gemm(src, w_pt1, B(94), hid, 8192, 512, 128, 1, nullptr, nullptr);
    gemm(hid, w_pt2, B(96), tA + (size_t)c * 8192 * DP, 8192, 128, 512, 0, nullptr, nullptr);
  }
  lnf(tA, bufPair, 91, 92, pairOut, LLi, DP);
}
